// MultiHeadAttention_57758720197285
// MI455X (gfx1250) — matrix operand order repaired and checked by reading, not compiled, not run on hardware
//
#include <hip/hip_runtime.h>
#include <hip/hip_bf16.h>

// ---------------------------------------------------------------------------
// MHA forward for MI455X (gfx1250): bf16 WMMA + Tensor Data Mover staging.
// B=4, L=2048, D=512, H=8, HD=64.  Causal mask applied analytically.
// ---------------------------------------------------------------------------

#define BB 4
#define LL 2048
#define DD 512
#define HH 8
#define HDD 64

typedef __bf16 bf16;
typedef bf16  v16bf __attribute__((ext_vector_type(16)));
typedef bf16  v8bf  __attribute__((ext_vector_type(8)));
typedef float v8f   __attribute__((ext_vector_type(8)));
typedef unsigned int u32x4 __attribute__((ext_vector_type(4)));
typedef int          i32x4 __attribute__((ext_vector_type(4)));
typedef int          i32x8 __attribute__((ext_vector_type(8)));

__device__ __forceinline__ v16bf mk16(v8bf lo, v8bf hi) {
    v16bf r;
#pragma unroll
    for (int i = 0; i < 8; ++i) { r[i] = lo[i]; r[i + 8] = hi[i]; }
    return r;
}

__device__ __forceinline__ v8f vzero8() {
    v8f z;
#pragma unroll
    for (int i = 0; i < 8; ++i) z[i] = 0.0f;
    return z;
}

// LDS byte offset of a __shared__ pointer (low 32 bits of the generic address).
__device__ __forceinline__ unsigned int lds_off(const void* p) {
    return (unsigned int)(uintptr_t)p;
}

// ---------------------------------------------------------------------------
// Tensor Data Mover: 2D tile load, row-major tensor -> contiguous LDS tile.
// Dimensions in elements (data_size = 2 bytes, bf16).
// ---------------------------------------------------------------------------
__device__ __forceinline__ void tdm_load_2d(const void* gaddr, unsigned int ldsoff,
                                            unsigned int tensor_d0, unsigned int tensor_d1,
                                            unsigned int stride0,
                                            unsigned int tile_d0, unsigned int tile_d1) {
    unsigned long long ga = (unsigned long long)(uintptr_t)gaddr;
    u32x4 g0;
    g0[0] = 1u;                                               // count=1, user desc
    g0[1] = ldsoff;                                           // lds_addr (bytes)
    g0[2] = (unsigned int)ga;                                 // global_addr[31:0]
    g0[3] = (unsigned int)((ga >> 32) & 0x1FFFFFFu) | (2u << 30); // addr[56:32] | type=2
    i32x8 g1;
    g1[0] = (int)(1u << 16);                                  // workgroup_mask=0, data_size=1 (2B)
    g1[1] = (int)((tensor_d0 & 0xFFFFu) << 16);               // tensor_dim0[15:0] @bits 63:48
    g1[2] = (int)(((tensor_d0 >> 16) & 0xFFFFu) |             // tensor_dim0[31:16]
                  ((tensor_d1 & 0xFFFFu) << 16));             // tensor_dim1[15:0]
    g1[3] = (int)(((tensor_d1 >> 16) & 0xFFFFu) |             // tensor_dim1[31:16]
                  ((tile_d0 & 0xFFFFu) << 16));               // tile_dim0
    g1[4] = (int)(tile_d1 & 0xFFFFu);                         // tile_dim1 (tile_dim2 = 0)
    g1[5] = (int)stride0;                                     // tensor_dim0_stride[31:0]
    g1[6] = 0;                                                // stride0 hi, dim1_stride lo
    g1[7] = 0;
    i32x4 z4 = {0, 0, 0, 0};
#if defined(__clang_major__) && (__clang_major__ >= 23)
    i32x8 z8 = {0, 0, 0, 0, 0, 0, 0, 0};
    __builtin_amdgcn_tensor_load_to_lds(g0, g1, z4, z4, z8, 0);
#else
    __builtin_amdgcn_tensor_load_to_lds(g0, g1, z4, z4, 0);
#endif
}

// ---------------------------------------------------------------------------
// Kernel 1: elementwise fp32 -> bf16 (vectorized x8)
// ---------------------------------------------------------------------------
__global__ void to_bf16(const float* __restrict__ s, bf16* __restrict__ d) {
    int i = (blockIdx.x * 256 + threadIdx.x) * 8;
    float4 f0 = *(const float4*)(s + i);
    float4 f1 = *(const float4*)(s + i + 4);
    v8bf o;
    o[0] = (bf16)f0.x; o[1] = (bf16)f0.y; o[2] = (bf16)f0.z; o[3] = (bf16)f0.w;
    o[4] = (bf16)f1.x; o[5] = (bf16)f1.y; o[6] = (bf16)f1.z; o[7] = (bf16)f1.w;
    *(v8bf*)(d + i) = o;
}

// ---------------------------------------------------------------------------
// Kernel 2: fused QKV projection GEMM.  out = x @ W^T + b.
// Q,K -> [B,H,L,HD] bf16.  V -> [B,H,HD,L] bf16 (pre-transposed for attention).
// Block = 128 threads (4 waves), 64x64 tile, K-step 32, TDM double-buffered.
// ---------------------------------------------------------------------------
__launch_bounds__(128)
__global__ void proj_gemm(const bf16* __restrict__ qx, const bf16* __restrict__ kx,
                          const bf16* __restrict__ vx,
                          const bf16* __restrict__ Wq, const bf16* __restrict__ Wk,
                          const bf16* __restrict__ Wv,
                          const float* __restrict__ bq, const float* __restrict__ bk,
                          const float* __restrict__ bv,
                          bf16* __restrict__ Qo, bf16* __restrict__ Ko,
                          bf16* __restrict__ Vo) {
    __shared__ __attribute__((aligned(32))) bf16 As[2][64][32];   // [m][k]
    __shared__ __attribute__((aligned(32))) bf16 Bs[2][64][32];   // [n][k] = W rows

    int z = blockIdx.z;
    const bf16*  x    = (z == 0) ? qx : (z == 1) ? kx : vx;
    const bf16*  W    = (z == 0) ? Wq : (z == 1) ? Wk : Wv;
    const float* bias = (z == 0) ? bq : (z == 1) ? bk : bv;
    bf16*        out  = (z == 0) ? Qo : (z == 1) ? Ko : Vo;

    int t = threadIdx.x;
    int wave = t >> 5, lane = t & 31, kh = lane >> 4, ln = lane & 15;
    int m0 = blockIdx.y * 64, n0 = blockIdx.x * 64;

    v8f acc[4];
#pragma unroll
    for (int i = 0; i < 4; ++i) acc[i] = vzero8();

    if (wave == 0) {  // stage k-tile 0
        tdm_load_2d(x + (size_t)m0 * DD + 0, lds_off(&As[0][0][0]), DD, BB * LL, DD, 32, 64);
        tdm_load_2d(W + (size_t)n0 * DD + 0, lds_off(&Bs[0][0][0]), DD, DD, DD, 32, 64);
    }

    for (int it = 0; it < DD / 32; ++it) {
        int cur = it & 1;
        if (wave == 0) __builtin_amdgcn_s_wait_tensorcnt(0);
        __syncthreads();
        if ((it + 1 < DD / 32) && wave == 0) {
            int k1 = (it + 1) * 32;
            tdm_load_2d(x + (size_t)m0 * DD + k1, lds_off(&As[cur ^ 1][0][0]), DD, BB * LL, DD, 32, 64);
            tdm_load_2d(W + (size_t)n0 * DD + k1, lds_off(&Bs[cur ^ 1][0][0]), DD, DD, DD, 32, 64);
        }
        int arow = wave * 16 + ln;
        v16bf a = mk16(*(const v8bf*)&As[cur][arow][16 * kh],
                       *(const v8bf*)&As[cur][arow][16 * kh + 8]);
#pragma unroll
        for (int nt = 0; nt < 4; ++nt) {
            const bf16* bp = &Bs[cur][nt * 16 + ln][16 * kh];
            v16bf b = mk16(*(const v8bf*)bp, *(const v8bf*)(bp + 8));
            acc[nt] = __builtin_amdgcn_wmma_f32_16x16x32_bf16(
                false, a, false, b, (short)0, acc[nt], false, false);
        }
    }

    // epilogue: + bias, bf16 store
    bool vtrans = (z == 2);
#pragma unroll
    for (int nt = 0; nt < 4; ++nt) {
        int col = n0 + nt * 16 + ln;
        float bb = bias[col];
        int h = col >> 6, hd = col & 63;
#pragma unroll
        for (int r = 0; r < 8; ++r) {
            int m = m0 + wave * 16 + r + 8 * kh;
            int b_ = m >> 11, l = m & (LL - 1);
            bf16 val = (bf16)(acc[nt][r] + bb);
            if (vtrans)
                out[(((size_t)(b_ * HH + h)) * HDD + hd) * LL + l] = val;   // [B,H,HD,L]
            else
                out[(((size_t)(b_ * HH + h)) * LL + l) * HDD + hd] = val;   // [B,H,L,HD]
        }
    }
}

// ---------------------------------------------------------------------------
// Kernel 3: flash attention, causal.  Block = 128 threads (4 waves).
// One (b,h) x 64 query rows per block; TDM double-buffered K / V^T tiles.
// ---------------------------------------------------------------------------
__launch_bounds__(128)
__global__ void flash_attn(const bf16* __restrict__ Q, const bf16* __restrict__ K,
                           const bf16* __restrict__ Vt, bf16* __restrict__ O) {
    __shared__ __attribute__((aligned(32))) bf16 Ks[2][64][64];   // [key][hd]
    __shared__ __attribute__((aligned(32))) bf16 Vs[2][64][64];   // [hd][key]
    __shared__ __attribute__((aligned(32))) bf16 Ps[4][16][64];   // per-wave P tile

    int t = threadIdx.x;
    int wave = t >> 5, lane = t & 31, kh = lane >> 4, ln = lane & 15;
    int q0 = blockIdx.x * 64;
    int bh = blockIdx.y;
    int ntiles = blockIdx.x + 1;

    const bf16* Qb  = Q  + ((size_t)bh * LL + q0) * HDD;
    const bf16* Kb  = K  + (size_t)bh * LL * HDD;
    const bf16* Vtb = Vt + (size_t)bh * HDD * LL;

    // Q rows for this wave, loaded straight into WMMA A layout (2 K-chunks).
    const bf16* qp = Qb + (size_t)(wave * 16 + ln) * HDD;
    v16bf aQ0 = mk16(*(const v8bf*)(qp + 16 * kh),      *(const v8bf*)(qp + 16 * kh + 8));
    v16bf aQ1 = mk16(*(const v8bf*)(qp + 32 + 16 * kh), *(const v8bf*)(qp + 32 + 16 * kh + 8));

    v8f ctx[4];
#pragma unroll
    for (int i = 0; i < 4; ++i) ctx[i] = vzero8();
    float mrow[8], lrow[8];
#pragma unroll
    for (int r = 0; r < 8; ++r) { mrow[r] = -3.0e38f; lrow[r] = 0.0f; }

    if (wave == 0) {  // stage tile 0
        tdm_load_2d(Kb,  lds_off(&Ks[0][0][0]), HDD, LL, HDD, 64, 64);
        tdm_load_2d(Vtb, lds_off(&Vs[0][0][0]), LL, HDD, LL, 64, 64);
    }

    for (int it = 0; it < ntiles; ++it) {
        int cur = it & 1;
        int j0 = it * 64;
        if (wave == 0) __builtin_amdgcn_s_wait_tensorcnt(0);
        __syncthreads();
        if ((it + 1 < ntiles) && wave == 0) {
            int j1 = j0 + 64;
            tdm_load_2d(Kb + (size_t)j1 * HDD, lds_off(&Ks[cur ^ 1][0][0]), HDD, LL, HDD, 64, 64);
            tdm_load_2d(Vtb + j1,              lds_off(&Vs[cur ^ 1][0][0]), LL, HDD, LL, 64, 64);
        }

        // ---- S = Q K^T ----
        v8f s[4];
#pragma unroll
        for (int nt = 0; nt < 4; ++nt) {
            const bf16* kp = &Ks[cur][nt * 16 + ln][0];
            v16bf b0 = mk16(*(const v8bf*)(kp + 16 * kh),      *(const v8bf*)(kp + 16 * kh + 8));
            v16bf b1 = mk16(*(const v8bf*)(kp + 32 + 16 * kh), *(const v8bf*)(kp + 32 + 16 * kh + 8));
            v8f sc = vzero8();
            sc = __builtin_amdgcn_wmma_f32_16x16x32_bf16(false, aQ0, false, b0, (short)0, sc, false, false);
            sc = __builtin_amdgcn_wmma_f32_16x16x32_bf16(false, aQ1, false, b1, (short)0, sc, false, false);
            s[nt] = sc;
        }

        // ---- scale + causal mask (diagonal tile only) ----
        bool diag = (j0 == q0);
#pragma unroll
        for (int nt = 0; nt < 4; ++nt)
#pragma unroll
            for (int r = 0; r < 8; ++r) {
                float val = s[nt][r] * 0.125f;   // 1/sqrt(64)
                if (diag) {
                    int col = j0 + nt * 16 + ln;
                    int row = q0 + wave * 16 + r + 8 * kh;
                    if (col > row) val = -1.0e9f;
                }
                s[nt][r] = val;
            }

        // ---- online softmax ----
#pragma unroll
        for (int r = 0; r < 8; ++r) {
            float mx = fmaxf(fmaxf(s[0][r], s[1][r]), fmaxf(s[2][r], s[3][r]));
            mx = fmaxf(mx, __shfl_xor(mx, 1, 32));
            mx = fmaxf(mx, __shfl_xor(mx, 2, 32));
            mx = fmaxf(mx, __shfl_xor(mx, 4, 32));
            mx = fmaxf(mx, __shfl_xor(mx, 8, 32));
            float nm = fmaxf(mrow[r], mx);
            float alpha = __expf(mrow[r] - nm);
            mrow[r] = nm;
            float sum = 0.0f;
#pragma unroll
            for (int nt = 0; nt < 4; ++nt) {
                float p = __expf(s[nt][r] - nm);
                s[nt][r] = p;
                sum += p;
            }
            sum += __shfl_xor(sum, 1, 32);
            sum += __shfl_xor(sum, 2, 32);
            sum += __shfl_xor(sum, 4, 32);
            sum += __shfl_xor(sum, 8, 32);
            lrow[r] = lrow[r] * alpha + sum;
#pragma unroll
            for (int nt = 0; nt < 4; ++nt) ctx[nt][r] *= alpha;
        }

        // ---- P: C layout -> A layout via per-wave LDS ----
#pragma unroll
        for (int nt = 0; nt < 4; ++nt)
#pragma unroll
            for (int r = 0; r < 8; ++r)
                Ps[wave][r + 8 * kh][nt * 16 + ln] = (bf16)s[nt][r];

        const bf16* pp = &Ps[wave][ln][0];
        v16bf aP0 = mk16(*(const v8bf*)(pp + 16 * kh),      *(const v8bf*)(pp + 16 * kh + 8));
        v16bf aP1 = mk16(*(const v8bf*)(pp + 32 + 16 * kh), *(const v8bf*)(pp + 32 + 16 * kh + 8));

        // ---- ctx += P V ----
#pragma unroll
        for (int nt = 0; nt < 4; ++nt) {
            const bf16* vp = &Vs[cur][nt * 16 + ln][0];
            v16bf b0 = mk16(*(const v8bf*)(vp + 16 * kh),      *(const v8bf*)(vp + 16 * kh + 8));
            v16bf b1 = mk16(*(const v8bf*)(vp + 32 + 16 * kh), *(const v8bf*)(vp + 32 + 16 * kh + 8));
            ctx[nt] = __builtin_amdgcn_wmma_f32_16x16x32_bf16(false, aP0, false, b0, (short)0, ctx[nt], false, false);
            ctx[nt] = __builtin_amdgcn_wmma_f32_16x16x32_bf16(false, aP1, false, b1, (short)0, ctx[nt], false, false);
        }
    }

    // ---- finalize ----
    float inv[8];
#pragma unroll
    for (int r = 0; r < 8; ++r) inv[r] = 1.0f / lrow[r];
    bf16* Ob = O + ((size_t)bh * LL + q0) * HDD;
#pragma unroll
    for (int nt = 0; nt < 4; ++nt)
#pragma unroll
        for (int r = 0; r < 8; ++r) {
            int row = wave * 16 + r + 8 * kh;
            Ob[(size_t)row * HDD + nt * 16 + ln] = (bf16)(ctx[nt][r] * inv[r]);
        }
}

// ---------------------------------------------------------------------------
// Kernel 4: output projection.  out = ctx @ Wo^T + bo, fp32 output [B,L,D].
// ctx is bf16 in [B,H,L,HD] layout.  TDM double-buffered.
// ---------------------------------------------------------------------------
__launch_bounds__(128)
__global__ void out_gemm(const bf16* __restrict__ ctx, const bf16* __restrict__ W,
                         const float* __restrict__ bias, float* __restrict__ out) {
    __shared__ __attribute__((aligned(32))) bf16 As[2][64][32];
    __shared__ __attribute__((aligned(32))) bf16 Bs[2][64][32];

    int t = threadIdx.x;
    int wave = t >> 5, lane = t & 31, kh = lane >> 4, ln = lane & 15;
    int m0 = blockIdx.y * 64, n0 = blockIdx.x * 64;
    int b_ = m0 >> 11, l0 = m0 & (LL - 1);

    v8f acc[4];
#pragma unroll
    for (int i = 0; i < 4; ++i) acc[i] = vzero8();

    // ctx A-tile base for k-chunk k0: head h = k0>>6, column offset k0&32.
    if (wave == 0) {
        const bf16* a0 = ctx + (((size_t)(b_ * HH + 0)) * LL + l0) * HDD + 0;
        tdm_load_2d(a0, lds_off(&As[0][0][0]), HDD, LL, HDD, 32, 64);
        tdm_load_2d(W + (size_t)n0 * DD + 0, lds_off(&Bs[0][0][0]), DD, DD, DD, 32, 64);
    }

    for (int it = 0; it < DD / 32; ++it) {
        int cur = it & 1;
        if (wave == 0) __builtin_amdgcn_s_wait_tensorcnt(0);
        __syncthreads();
        if ((it + 1 < DD / 32) && wave == 0) {
            int k1 = (it + 1) * 32;
            int h = k1 >> 6, off = k1 & 32;
            const bf16* a1 = ctx + (((size_t)(b_ * HH + h)) * LL + l0) * HDD + off;
            tdm_load_2d(a1, lds_off(&As[cur ^ 1][0][0]), HDD, LL, HDD, 32, 64);
            tdm_load_2d(W + (size_t)n0 * DD + k1, lds_off(&Bs[cur ^ 1][0][0]), DD, DD, DD, 32, 64);
        }
        int arow = wave * 16 + ln;
        v16bf a = mk16(*(const v8bf*)&As[cur][arow][16 * kh],
                       *(const v8bf*)&As[cur][arow][16 * kh + 8]);
#pragma unroll
        for (int nt = 0; nt < 4; ++nt) {
            const bf16* bp = &Bs[cur][nt * 16 + ln][16 * kh];
            v16bf b = mk16(*(const v8bf*)bp, *(const v8bf*)(bp + 8));
            acc[nt] = __builtin_amdgcn_wmma_f32_16x16x32_bf16(
                false, a, false, b, (short)0, acc[nt], false, false);
        }
    }

#pragma unroll
    for (int nt = 0; nt < 4; ++nt) {
        int col = n0 + nt * 16 + ln;
        float bb = bias[col];
#pragma unroll
        for (int r = 0; r < 8; ++r) {
            int m = m0 + wave * 16 + r + 8 * kh;
            out[(size_t)m * DD + col] = acc[nt][r] + bb;
        }
    }
}

// ---------------------------------------------------------------------------
// Launch
// ---------------------------------------------------------------------------
extern "C" void kernel_launch(void* const* d_in, const int* in_sizes, int n_in,
                              void* d_out, int out_size, void* d_ws, size_t ws_size,
                              hipStream_t stream) {
    const float* q  = (const float*)d_in[0];
    const float* k  = (const float*)d_in[1];
    const float* v  = (const float*)d_in[2];
    // d_in[3] = mask (tril) — applied analytically in flash_attn
    const float* Wq = (const float*)d_in[4];
    const float* bq = (const float*)d_in[5];
    const float* Wk = (const float*)d_in[6];
    const float* bk = (const float*)d_in[7];
    const float* Wv = (const float*)d_in[8];
    const float* bv = (const float*)d_in[9];
    const float* Wo = (const float*)d_in[10];
    const float* bo = (const float*)d_in[11];

    char* ws = (char*)d_ws;
    size_t off = 0;
    auto alloc = [&](size_t bytes) -> void* {
        void* p = ws + off;
        off = (off + bytes + 255) & ~(size_t)255;
        return p;
    };
    const size_t wn = (size_t)DD * DD;          // weight elements
    const size_t an = (size_t)BB * LL * DD;     // activation elements
    bf16* Wqb = (bf16*)alloc(wn * 2);
    bf16* Wkb = (bf16*)alloc(wn * 2);
    bf16* Wvb = (bf16*)alloc(wn * 2);
    bf16* Wob = (bf16*)alloc(wn * 2);
    bf16* xq  = (bf16*)alloc(an * 2);
    bf16* xk  = (bf16*)alloc(an * 2);
    bf16* xv  = (bf16*)alloc(an * 2);
    bf16* Qb  = (bf16*)alloc(an * 2);
    bf16* Kb  = (bf16*)alloc(an * 2);
    bf16* Vtb = (bf16*)alloc(an * 2);
    bf16* Cb  = (bf16*)alloc(an * 2);

    // fp32 -> bf16 conversions (weights stay [N][K]: already B-operand layout)
    to_bf16<<<wn / (256 * 8), 256, 0, stream>>>(Wq, Wqb);
    to_bf16<<<wn / (256 * 8), 256, 0, stream>>>(Wk, Wkb);
    to_bf16<<<wn / (256 * 8), 256, 0, stream>>>(Wv, Wvb);
    to_bf16<<<wn / (256 * 8), 256, 0, stream>>>(Wo, Wob);
    to_bf16<<<an / (256 * 8), 256, 0, stream>>>(q, xq);
    to_bf16<<<an / (256 * 8), 256, 0, stream>>>(k, xk);
    to_bf16<<<an / (256 * 8), 256, 0, stream>>>(v, xv);

    proj_gemm<<<dim3(DD / 64, (BB * LL) / 64, 3), 128, 0, stream>>>(
        xq, xk, xv, Wqb, Wkb, Wvb, bq, bk, bv, Qb, Kb, Vtb);

    flash_attn<<<dim3(LL / 64, BB * HH), 128, 0, stream>>>(Qb, Kb, Vtb, Cb);

    out_gemm<<<dim3(DD / 64, (BB * LL) / 64), 128, 0, stream>>>(
        Cb, Wob, bo, (float*)d_out);
}
